// MultiEdgeConvolution_33827162423732
// MI455X (gfx1250) — compile-verified, matrix operand order checked by reading, not run on hardware
//
#include <hip/hip_runtime.h>
#include <hip/hip_bf16.h>

// ---------------------------------------------------------------------------
// DGCNN edge-conv block, fused for MI455X (gfx1250, wave32, WMMA).
//   B=16, C=3, N=4096, K=20, ch0=6, ch1=64, ch2=64
// Pipeline:
//   k_init    : zero small accumulators in ws
//   k_knn     : brute-force top-20 KNN per point (x[b] in LDS)
//   k_stats1  : 6-vector sum + 6x6 second moment of edge features
//               -> exact BN1 stats via linearity of conv1
//   k_fin1    : fold BN1 into per-channel scale/bias
//   k_main    : fused gather + conv1 + BN1 + lrelu -> f16 LDS tile,
//               conv2 as h2^T = h1^T * W1^T via v_wmma_f32_16x16x32_f16
//               (k-dim padded onto the M axis so max/min over k is done with
//                register v_max/v_min + one shfl_xor, not per-element atomics)
//   k_fin2    : fold BN2 into scale/bias
//   k_out     : out = lrelu(scale2 * (scale2>=0 ? max_k : min_k) + bias2)
// ---------------------------------------------------------------------------

typedef __attribute__((ext_vector_type(16))) _Float16 v16h;
typedef __attribute__((ext_vector_type(8)))  float    v8f;

#define NPTS   4096
#define NBATCH 16
#define KNN    20
#define NCOLS  1310720      /* B*N*K */
#define LRELU  0.2f

__device__ __forceinline__ unsigned f2ord(float f) {
  unsigned u = __float_as_uint(f);
  return (u & 0x80000000u) ? ~u : (u | 0x80000000u);
}
__device__ __forceinline__ float ord2f(unsigned o) {
  unsigned u = (o & 0x80000000u) ? (o & 0x7fffffffu) : ~o;
  return __uint_as_float(u);
}

// ---------------------------------------------------------------- init ------
__global__ void k_init(float* smallws) {
  int i = blockIdx.x * 256 + threadIdx.x;   // 512 floats
  if (i < 512) smallws[i] = 0.f;
}

// ----------------------------------------------------------------- knn ------
// 256 blocks (16 batches x 16 chunks of 256 points), 256 threads.
__global__ void k_knn(const float* __restrict__ x, int* __restrict__ idx) {
  __shared__ float sx[3 * NPTS];            // 48 KB: x[b] staged in LDS
  int b     = blockIdx.x >> 4;
  int chunk = blockIdx.x & 15;
  const float* xb = x + (size_t)b * 3 * NPTS;
  for (int i = threadIdx.x; i < 3 * NPTS; i += 256) sx[i] = xb[i];
  __syncthreads();

  int n = chunk * 256 + threadIdx.x;
  float xn0 = sx[n], xn1 = sx[NPTS + n], xn2 = sx[2 * NPTS + n];
  float xxn = xn0 * xn0 + xn1 * xn1 + xn2 * xn2;

  float bd[KNN]; int bi[KNN];
#pragma unroll
  for (int j = 0; j < KNN; ++j) { bd[j] = -3.4e38f; bi[j] = 0; }

  for (int m = 0; m < NPTS; ++m) {
    float m0 = sx[m], m1 = sx[NPTS + m], m2 = sx[2 * NPTS + m];
    float inner = xn0 * m0 + xn1 * m1 + xn2 * m2;
    float xxm   = m0 * m0 + m1 * m1 + m2 * m2;
    float nd    = 2.f * inner - xxn - xxm;    // neg squared distance
    if (nd > bd[KNN - 1]) {
      bd[KNN - 1] = nd; bi[KNN - 1] = m;
#pragma unroll
      for (int j = KNN - 1; j > 0; --j) {
        if (bd[j] > bd[j - 1]) {
          float t = bd[j]; bd[j] = bd[j - 1]; bd[j - 1] = t;
          int  ti = bi[j]; bi[j] = bi[j - 1]; bi[j - 1] = ti;
        }
      }
    }
  }
  int* op = idx + ((size_t)b * NPTS + n) * KNN;
#pragma unroll
  for (int j = 0; j < KNN; ++j) op[j] = bi[j];
}

// --------------------------------------------------------------- stats1 -----
// Edge-feature first/second moments: s[6] and upper-tri G[21].
__global__ void k_stats1(const float* __restrict__ x, const int* __restrict__ idx,
                         float* __restrict__ acc1) {
  __shared__ float la[32];
  if (threadIdx.x < 32) la[threadIdx.x] = 0.f;
  __syncthreads();

  int g = blockIdx.x * 256 + threadIdx.x;   // 0 .. 65535 == (b,n)
  int b = g >> 12, n = g & (NPTS - 1);
  const float* xb = x + (size_t)b * 3 * NPTS;
  float x0 = xb[n], x1 = xb[NPTS + n], x2 = xb[2 * NPTS + n];
  const int* ip = idx + (size_t)g * KNN;

  float s[6] = {0, 0, 0, 0, 0, 0};
  float G[21];
#pragma unroll
  for (int t = 0; t < 21; ++t) G[t] = 0.f;

  for (int k = 0; k < KNN; ++k) {
    int m = ip[k];
    float e[6];
    e[0] = xb[m] - x0; e[1] = xb[NPTS + m] - x1; e[2] = xb[2 * NPTS + m] - x2;
    e[3] = x0; e[4] = x1; e[5] = x2;
    int t = 0;
#pragma unroll
    for (int a = 0; a < 6; ++a) {
      s[a] += e[a];
#pragma unroll
      for (int c2 = a; c2 < 6; ++c2) { G[t] += e[a] * e[c2]; ++t; }
    }
  }
#pragma unroll
  for (int a = 0; a < 6; ++a) atomicAdd(&la[a], s[a]);
#pragma unroll
  for (int t = 0; t < 21; ++t) atomicAdd(&la[6 + t], G[t]);
  __syncthreads();
  if (threadIdx.x < 27) atomicAdd(&acc1[threadIdx.x], la[threadIdx.x]);
}

// ----------------------------------------------------------------- fin1 -----
__global__ void k_fin1(const float* __restrict__ W0, const float* __restrict__ g0,
                       const float* __restrict__ b0, const float* __restrict__ acc1,
                       float* __restrict__ p0) {
  int c = threadIdx.x;                      // 64
  const float invM = 1.f / (float)NCOLS;
  float w[6];
#pragma unroll
  for (int j = 0; j < 6; ++j) w[j] = W0[c * 6 + j];
  float m1 = 0.f;
#pragma unroll
  for (int j = 0; j < 6; ++j) m1 += w[j] * acc1[j];
  m1 *= invM;
  float ex2 = 0.f; int t = 0;
#pragma unroll
  for (int a = 0; a < 6; ++a)
#pragma unroll
    for (int c2 = a; c2 < 6; ++c2) {
      float coef = (a == c2) ? 1.f : 2.f;
      ex2 += coef * w[a] * w[c2] * acc1[6 + t]; ++t;
    }
  ex2 *= invM;
  float var = ex2 - m1 * m1;
  float sc  = g0[c] * rsqrtf(var + 1e-5f);
  p0[c] = sc;
  p0[64 + c] = b0[c] - m1 * sc;
}

// ----------------------------------------------------------------- main -----
// 4096 blocks x 256 threads; block = (batch b, 16 points) -> 320 real columns,
// padded to 16 n x 32 k on the WMMA M axis (pad cols clamp to k=19).
__global__ void k_main(const float* __restrict__ x, const int* __restrict__ idx,
                       const float* __restrict__ W0g, const float* __restrict__ W1g,
                       const float* __restrict__ p0, float* __restrict__ acc2,
                       float* __restrict__ maxg, float* __restrict__ ming) {
  __shared__ _Float16 h1[320 * 64];         // 40 KB f16 activation tile [col][ch]
  __shared__ float    sW0[64 * 6];
  __shared__ float    ssc[64], sbi[64];
  __shared__ unsigned mx[64 * 16], mn[64 * 16];
  __shared__ float    ssum[64], ssq[64];

  int tid   = threadIdx.x;
  int b     = blockIdx.x >> 8;              // 256 blocks per batch
  int nbase = (blockIdx.x & 255) * 16;

  for (int i = tid; i < 384; i += 256) sW0[i] = W0g[i];
  if (tid < 64) { ssc[tid] = p0[tid]; sbi[tid] = p0[64 + tid]; ssum[tid] = 0.f; ssq[tid] = 0.f; }
  for (int i = tid; i < 1024; i += 256) { mx[i] = 0u; mn[i] = 0xFFFFFFFFu; }
  __syncthreads();

  const float* xb = x + (size_t)b * 3 * NPTS;

  // ---- Phase A: gather + conv1 + BN1 + lrelu -> f16 LDS (col-major, 64/col)
  for (int l = tid; l < 320; l += 256) {
    int nl = l / 20, k = l % 20;
    int n  = nbase + nl;
    int m  = idx[(((size_t)b * NPTS) + n) * KNN + k];
    float x0 = xb[n],            x1 = xb[NPTS + n],      x2 = xb[2 * NPTS + n];
    float e0 = xb[m] - x0,       e1 = xb[NPTS + m] - x1, e2 = xb[2 * NPTS + m] - x2;
    _Float16* hp = &h1[l * 64];
#pragma unroll 4
    for (int c = 0; c < 64; ++c) {
      const float* w = &sW0[c * 6];
      float v = w[0] * e0 + w[1] * e1 + w[2] * e2 + w[3] * x0 + w[4] * x1 + w[5] * x2;
      v = v * ssc[c] + sbi[c];
      v = (v >= 0.f) ? v : LRELU * v;
      hp[c] = (_Float16)v;
    }
  }
  __syncthreads();

  // ---- Phase B: h2^T = h1^T(512x64, k-padded) * W1^T(64x64) via WMMA -----
  // Tile: M = 16 k-slots of ONE point n, N = 16 out-channels, K = 64 (2x32).
  int wv     = tid >> 5;                    // 8 waves
  int lane   = tid & 31;
  int r      = wv & 3;                      // channel (N) tile 0..3
  int mgroup = wv >> 2;                     // M-tile group: 16 tiles each
  int hl     = lane & 15;
  int kb     = (lane < 16) ? 0 : 8;         // 16-bit A/B lane striping (ISA 7.12.2)
  int ch     = 16 * r + hl;                 // this lane's output channel

  // B-fragment = W1^T chunk, fixed per wave: B[kk][n'] = W1[16r+n'][kk]
  v16h Bf[2];
#pragma unroll
  for (int q = 0; q < 2; ++q) {
    const float* wrow = W1g + (size_t)ch * 64 + 32 * q;
#pragma unroll
    for (int e = 0; e < 16; ++e) {
      int kk = (e < 8) ? (kb + e) : (16 + kb + (e - 8));
      Bf[q][e] = (_Float16)wrow[kk];
    }
  }

  float ssum_l = 0.f, ssq_l = 0.f;
  float rmax = -3.4e38f, rmin = 3.4e38f;

  for (int t = 0; t < 16; ++t) {
    int mt    = mgroup * 16 + t;            // M-tile 0..31
    int nl    = mt >> 1;                    // point within block (0..15)
    int khalf = mt & 1;                     // k 0..15 or 16..31(padded)

    // A-fragment: A[m][kdim] = h1[col(nl, khalf*16+m)][kdim]
    int kp = khalf * 16 + hl;               // virtual k slot of this lane
    int kc = (kp > 19) ? 19 : kp;           // pad cols duplicate k=19
    const _Float16* acol = &h1[(nl * 20 + kc) * 64];
    v16h Af[2];
#pragma unroll
    for (int q = 0; q < 2; ++q) {
#pragma unroll
      for (int e = 0; e < 16; ++e) {
        int kk = (e < 8) ? (kb + e) : (16 + kb + (e - 8));
        Af[q][e] = acol[32 * q + kk];
      }
    }
    v8f acc = {};
    acc = __builtin_amdgcn_wmma_f32_16x16x32_f16(false, Af[0], false, Bf[0], (short)0, acc, false, false);
    acc = __builtin_amdgcn_wmma_f32_16x16x32_f16(false, Af[1], false, Bf[1], (short)0, acc, false, false);

    // ---- Phase C: register max/min over the 8 k's this lane holds -------
    float tmax = acc[0], tmin = acc[0];
#pragma unroll
    for (int j = 1; j < 8; ++j) { tmax = fmaxf(tmax, acc[j]); tmin = fminf(tmin, acc[j]); }

    if (khalf == 0) {                       // wave-uniform branch
      rmax = tmax; rmin = tmin;
      // k = j + kb in [0,15]: all 8 valid for stats
#pragma unroll
      for (int j = 0; j < 8; ++j) { float v = acc[j]; ssum_l += v; ssq_l += v * v; }
    } else {
      rmax = fmaxf(rmax, tmax); rmin = fminf(rmin, tmin);
      if (lane < 16) {                      // k = 16+j valid only for j<4
#pragma unroll
        for (int j = 0; j < 4; ++j) { float v = acc[j]; ssum_l += v; ssq_l += v * v; }
      }
      // merge half-wave partner (other 8 k-slots of same (n, ch))
      float om = __shfl_xor(rmax, 16, 32);
      float on = __shfl_xor(rmin, 16, 32);
      rmax = fmaxf(rmax, om); rmin = fminf(rmin, on);
      if (lane < 16) {
        atomicMax(&mx[ch * 16 + nl], f2ord(rmax));
        atomicMin(&mn[ch * 16 + nl], f2ord(rmin));
      }
    }
  }
  atomicAdd(&ssum[ch], ssum_l);
  atomicAdd(&ssq[ch],  ssq_l);
  __syncthreads();

  if (tid < 64) {
    atomicAdd(&acc2[tid],      ssum[tid]);
    atomicAdd(&acc2[64 + tid], ssq[tid]);
    size_t obase = ((size_t)b * 64 + tid) * NPTS + nbase;
#pragma unroll
    for (int nl = 0; nl < 16; ++nl) {
      maxg[obase + nl] = ord2f(mx[tid * 16 + nl]);
      ming[obase + nl] = ord2f(mn[tid * 16 + nl]);
    }
  }
}

// ----------------------------------------------------------------- fin2 -----
__global__ void k_fin2(const float* __restrict__ acc2, const float* __restrict__ g1,
                       const float* __restrict__ b1, float* __restrict__ p2) {
  int c = threadIdx.x;                      // 64
  const float invM = 1.f / (float)NCOLS;
  float m1  = acc2[c] * invM;
  float ex2 = acc2[64 + c] * invM;
  float var = ex2 - m1 * m1;
  float sc  = g1[c] * rsqrtf(var + 1e-5f);
  p2[c] = sc;
  p2[64 + c] = b1[c] - m1 * sc;
}

// ------------------------------------------------------------------ out -----
__global__ void k_out(const float* __restrict__ maxg, const float* __restrict__ ming,
                      const float* __restrict__ p2, float* __restrict__ out) {
  size_t i = (size_t)blockIdx.x * 256 + threadIdx.x;   // 16*64*4096 elements
  int c = (int)((i >> 12) & 63);
  float a = p2[c], bb = p2[64 + c];
  float h = (a >= 0.f) ? maxg[i] : ming[i];   // max commutes through monotone BN+lrelu
  float v = a * h + bb;
  out[i] = (v >= 0.f) ? v : LRELU * v;
}

// --------------------------------------------------------------- launch -----
extern "C" void kernel_launch(void* const* d_in, const int* in_sizes, int n_in,
                              void* d_out, int out_size, void* d_ws, size_t ws_size,
                              hipStream_t stream) {
  const float* x  = (const float*)d_in[0];
  const float* W0 = (const float*)d_in[1];
  const float* g0 = (const float*)d_in[2];
  const float* b0 = (const float*)d_in[3];
  const float* W1 = (const float*)d_in[4];
  const float* g1 = (const float*)d_in[5];
  const float* b1 = (const float*)d_in[6];
  float* out = (float*)d_out;

  // workspace layout
  int*   d_idx   = (int*)d_ws;                                   // 5,242,880 B
  float* d_small = (float*)((char*)d_ws + 5242880);              // 512 floats
  float* d_acc1  = d_small + 0;      // 27 used
  float* d_p0    = d_small + 128;    // scale0[64], bias0[64]
  float* d_acc2  = d_small + 256;    // sum[64], sumsq[64]
  float* d_p2    = d_small + 384;    // scale2[64], bias2[64]
  float* d_max   = (float*)((char*)d_ws + 5242880 + 4096);       // 16,777,216 B
  float* d_min   = d_max + (size_t)NBATCH * 64 * NPTS;           // 16,777,216 B
  (void)in_sizes; (void)n_in; (void)out_size; (void)ws_size;

  k_init  <<<2,     256, 0, stream>>>(d_small);
  k_knn   <<<256,   256, 0, stream>>>(x, d_idx);
  k_stats1<<<256,   256, 0, stream>>>(x, d_idx, d_acc1);
  k_fin1  <<<1,      64, 0, stream>>>(W0, g0, b0, d_acc1, d_p0);
  k_main  <<<4096,  256, 0, stream>>>(x, d_idx, W0, W1, d_p0, d_acc2, d_max, d_min);
  k_fin2  <<<1,      64, 0, stream>>>(d_acc2, g1, b1, d_p2);
  k_out   <<<16384, 256, 0, stream>>>(d_max, d_min, d_p2, out);
}
